// DrugLAMPBase_87265145520699
// MI455X (gfx1250) — compile-verified, matrix operand order checked by reading, not run on hardware
//
#include <hip/hip_runtime.h>
#include <math.h>

typedef _Float16 v16h __attribute__((ext_vector_type(16)));
typedef _Float16 v8h  __attribute__((ext_vector_type(8)));
typedef _Float16 v4h  __attribute__((ext_vector_type(4)));
typedef float    v8f  __attribute__((ext_vector_type(8)));

__device__ __forceinline__ float gelu_erf(float x) {
  return 0.5f * x * (1.0f + erff(x * 0.70710678118654752f));
}

// ---------------- utility kernels ----------------
__global__ void zero_f32(float* __restrict__ p, long n) {
  long i = (long)blockIdx.x * blockDim.x + threadIdx.x;
  long stride = (long)gridDim.x * blockDim.x;
  for (; i < n; i += stride) p[i] = 0.f;
}

__global__ void degree_kernel(const int* __restrict__ src, const int* __restrict__ dst,
                              float* __restrict__ outdeg, float* __restrict__ indeg, long E) {
  long i = (long)blockIdx.x * blockDim.x + threadIdx.x;
  if (i < E) {
    atomicAdd(&outdeg[src[i]], 1.0f);
    atomicAdd(&indeg[dst[i]], 1.0f);
  }
}

__global__ void rsqrt_clip_kernel(float* __restrict__ p, long n) {
  long i = (long)blockIdx.x * blockDim.x + threadIdx.x;
  if (i < n) p[i] = rsqrtf(fmaxf(p[i], 1.0f));
}

// h[i,:] = x[i,:] * ro[i]  (f32 in, f16 out)
__global__ void scale_rows_kernel(const float* __restrict__ x, const float* __restrict__ ro,
                                  _Float16* __restrict__ h, long nelem) {
  long i = (long)blockIdx.x * blockDim.x + threadIdx.x;
  if (i < nelem) h[i] = (_Float16)(x[i] * ro[i >> 7]);
}

// f32 -> f16 elementwise
__global__ void cvt_f32_f16_kernel(const float* __restrict__ s, _Float16* __restrict__ d, long n) {
  long i = (long)blockIdx.x * blockDim.x + threadIdx.x;
  if (i < n) d[i] = (_Float16)s[i];
}

// wt[o*Kin + k] = (f16) w[k*Kout + o]   (transpose + convert; small, one-shot)
__global__ void wtrans_kernel(const float* __restrict__ w, _Float16* __restrict__ wt,
                              int Kin, int Kout) {
  long i = (long)blockIdx.x * blockDim.x + threadIdx.x;
  if (i < (long)Kin * Kout) {
    int o = (int)(i / Kin), k = (int)(i % Kin);
    wt[i] = (_Float16)w[(long)k * Kout + o];
  }
}

// agg[dst[e], :] += h[src[e], :]   (32 lanes per edge, 4 halves each; f32 atomics)
__global__ void scatter_kernel(const _Float16* __restrict__ h, const int* __restrict__ src,
                               const int* __restrict__ dst, float* __restrict__ agg, long E) {
  long gid = (long)blockIdx.x * blockDim.x + threadIdx.x;
  long e = gid >> 5;
  if (e >= E) return;
  int c = (int)(gid & 31) * 4;
  int s = src[e], d = dst[e];
  v4h v = *(const v4h*)(h + (long)s * 128 + c);
  float* o = agg + (long)d * 128 + c;
  atomicAdd(o + 0, (float)v[0]);
  atomicAdd(o + 1, (float)v[1]);
  atomicAdd(o + 2, (float)v[2]);
  atomicAdd(o + 3, (float)v[3]);
}

// ---------------- WMMA GEMM: out = epi(A[M,Kin] @ W + bias), all-f16 data path ----------
enum { EPI_LIN = 0, EPI_GELU = 1, EPI_GCN = 2 };

#define LDT 40              // LDS tile row stride in halves (80 B)
#define SZT (128 * LDT)     // halves per tile buffer

template <int MODE>
__global__ __launch_bounds__(256) void gemm_wmma_kernel(
    const _Float16* __restrict__ A,   // [M x Kin] f16
    const _Float16* __restrict__ Wt,  // [Kout x Kin] f16 (pre-transposed)
    const float* __restrict__ bias, const float* __restrict__ rowscale,
    _Float16* __restrict__ out,       // [M x Kout] f16
    int M, int Kin, int Kout) {
  // Layout: [As0 | As1 | Bs0 | Bs1]; epilogue reuses the front as a 128x128 C tile.
  __shared__ __align__(16) _Float16 smem[4 * SZT];

  const int t = threadIdx.x;
  const int lane = t & 31, wid = t >> 5;
  const int wm = wid >> 1, wn = wid & 1;   // 4 x 2 wave grid -> 32x64 per wave
  const int g = lane >> 4, l16 = lane & 15;
  const int m0 = blockIdx.y * 128;
  const int n0 = blockIdx.x * 128;
  const int nk = Kin >> 5;

  // Branch-free staging: rows >= M are clamped to M-1 (their outputs are never stored).
  auto stageA = [&](int k0, int base) {
#pragma unroll
    for (int i = 0; i < 2; ++i) {
      int c = i * 256 + t;        // 512 chunks of 8 halves
      int r = c >> 2;             // 0..127
      int kc = (c & 3) << 3;      // 0,8,16,24
      int row = m0 + r;
      int rowc = row < M ? row : (M - 1);
      *(v8h*)&smem[base + r * LDT + kc] = *(const v8h*)(A + (long)rowc * Kin + k0 + kc);
    }
  };
  auto stageB = [&](int k0, int base) {
#pragma unroll
    for (int i = 0; i < 2; ++i) {
      int c = i * 256 + t;
      int n = c >> 2;
      int kc = (c & 3) << 3;
      *(v8h*)&smem[base + n * LDT + kc] = *(const v8h*)(Wt + (long)(n0 + n) * Kin + k0 + kc);
    }
  };

  v8f acc[2][4] = {};
  stageA(0, 0);
  stageB(0, 2 * SZT);
  __syncthreads();

  for (int kt = 0; kt < nk; ++kt) {
    int p = kt & 1;
    if (kt + 1 < nk) {               // prefetch next K-tile into the other buffer
      stageA((kt + 1) * 32, (p ^ 1) * SZT);
      stageB((kt + 1) * 32, 2 * SZT + (p ^ 1) * SZT);
    }
    const int aO = p * SZT;
    const int bO = 2 * SZT + p * SZT;

    union F { v16h v; v8h h[2]; } a[2], b[4];
#pragma unroll
    for (int ms = 0; ms < 2; ++ms) {
      int row = wm * 32 + ms * 16 + l16;
      a[ms].h[0] = *(const v8h*)&smem[aO + row * LDT + g * 8];       // K = g*8..g*8+7
      a[ms].h[1] = *(const v8h*)&smem[aO + row * LDT + 16 + g * 8];  // K = 16+g*8..
    }
#pragma unroll
    for (int ns = 0; ns < 4; ++ns) {
      int col = wn * 64 + ns * 16 + l16;
      b[ns].h[0] = *(const v8h*)&smem[bO + col * LDT + g * 8];
      b[ns].h[1] = *(const v8h*)&smem[bO + col * LDT + 16 + g * 8];
    }
#pragma unroll
    for (int ms = 0; ms < 2; ++ms)
#pragma unroll
      for (int ns = 0; ns < 4; ++ns)
        acc[ms][ns] = __builtin_amdgcn_wmma_f32_16x16x32_f16(
            false, a[ms].v, false, b[ns].v, (short)0, acc[ms][ns], false, false);
    __syncthreads();
  }

  // Epilogue: apply per-element math, bounce tile through LDS, store coalesced b128.
  // Cs occupies smem[0 .. 16384) (16384 <= 4*SZT = 20480).
#pragma unroll
  for (int ms = 0; ms < 2; ++ms) {
#pragma unroll
    for (int ns = 0; ns < 4; ++ns) {
      int colL = wn * 64 + ns * 16 + l16;
#pragma unroll
      for (int r = 0; r < 8; ++r) {
        int rowL = wm * 32 + ms * 16 + g * 8 + r;
        float v = acc[ms][ns][r];
        if (MODE == EPI_GCN) {
          int row = m0 + rowL;
          int rc = row < M ? row : (M - 1);
          v = v * rowscale[rc] + bias[n0 + colL];
        }
        if (MODE == EPI_GELU) v = gelu_erf(v + bias[n0 + colL]);
        if (MODE == EPI_LIN)  v = v + bias[n0 + colL];
        smem[rowL * 128 + colL] = (_Float16)v;
      }
    }
  }
  __syncthreads();
#pragma unroll
  for (int i = 0; i < 8; ++i) {
    int c = i * 256 + t;            // 2048 chunks of 8 halves
    int r = c >> 4;                 // 0..127
    int cc = (c & 15) << 3;         // 0..120
    int row = m0 + r;
    if (row < M)
      *(v8h*)(out + (long)row * Kout + n0 + cc) = *(const v8h*)&smem[r * 128 + cc];
  }
}

// ---------------- LayerNorm over 512 cols (wave per row, f16 data) ----------------
__global__ void layernorm_kernel(_Float16* __restrict__ f, const float* __restrict__ gamma,
                                 const float* __restrict__ beta, int n) {
  int gw = (int)(((long)blockIdx.x * blockDim.x + threadIdx.x) >> 5);
  int lane = threadIdx.x & 31;
  if (gw >= n) return;
  _Float16* row = f + (long)gw * 512;
  v8h u0 = *(const v8h*)(row + lane * 16);
  v8h u1 = *(const v8h*)(row + lane * 16 + 8);
  float v[16];
  float s = 0.f, q = 0.f;
#pragma unroll
  for (int i = 0; i < 8; ++i) {
    v[i] = (float)u0[i];
    v[8 + i] = (float)u1[i];
  }
#pragma unroll
  for (int i = 0; i < 16; ++i) { s += v[i]; q += v[i] * v[i]; }
#pragma unroll
  for (int o = 16; o > 0; o >>= 1) { s += __shfl_xor(s, o); q += __shfl_xor(q, o); }
  float mean = s * (1.f / 512.f);
  float var = q * (1.f / 512.f) - mean * mean;
  float rstd = rsqrtf(var + 1e-5f);
  v8h w0, w1;
#pragma unroll
  for (int i = 0; i < 8; ++i) {
    int c0 = lane * 16 + i, c1 = lane * 16 + 8 + i;
    w0[i] = (_Float16)((v[i] - mean) * rstd * gamma[c0] + beta[c0]);
    w1[i] = (_Float16)((v[8 + i] - mean) * rstd * gamma[c1] + beta[c1]);
  }
  *(v8h*)(row + lane * 16) = w0;
  *(v8h*)(row + lane * 16 + 8) = w1;
}

// ---------------- BatchNorm: column sums / sumsq then apply (f16 data) ----------------
__global__ void bn_stats_kernel(const _Float16* __restrict__ x, float* __restrict__ sums,
                                int n, int C) {
  const int ROWS = 256;
  int r0 = blockIdx.x * ROWS;
  int t = threadIdx.x;
  float s0 = 0.f, q0 = 0.f, s1 = 0.f, q1 = 0.f;
  int r1 = min(n, r0 + ROWS);
  for (int r = r0; r < r1; ++r) {
    const _Float16* row = x + (long)r * C;
    if (t < C) { float v = (float)row[t]; s0 += v; q0 += v * v; }
    if (C > 256) { float v = (float)row[t + 256]; s1 += v; q1 += v * v; }
  }
  if (t < C) { atomicAdd(&sums[t], s0); atomicAdd(&sums[C + t], q0); }
  if (C > 256) { atomicAdd(&sums[t + 256], s1); atomicAdd(&sums[C + t + 256], q1); }
}

__global__ void bn_apply_kernel(_Float16* __restrict__ x, const float* __restrict__ sums,
                                const float* __restrict__ g, const float* __restrict__ b,
                                long n, int C) {
  long i = (long)blockIdx.x * blockDim.x + threadIdx.x;
  if (i >= n * C) return;
  int c = (int)(i % C);
  float inv_n = 1.0f / (float)n;
  float m = sums[c] * inv_n;
  float var = sums[C + c] * inv_n - m * m;
  x[i] = (_Float16)(((float)x[i] - m) * rsqrtf(var + 1e-5f) * g[c] + b[c]);
}

// ---------------- final fc4: [N,128] @ [128,1] (wave per row) ----------------
__global__ void fc4_kernel(const _Float16* __restrict__ h, const float* __restrict__ w,
                           const float* __restrict__ b, float* __restrict__ out, int n) {
  int gw = (int)(((long)blockIdx.x * blockDim.x + threadIdx.x) >> 5);
  int lane = threadIdx.x & 31;
  if (gw >= n) return;
  const _Float16* row = h + (long)gw * 128;
  float s = 0.f;
#pragma unroll
  for (int i = 0; i < 4; ++i) s += (float)row[lane + i * 32] * w[lane + i * 32];
#pragma unroll
  for (int o = 16; o > 0; o >>= 1) s += __shfl_xor(s, o);
  if (lane == 0) out[gw] = s + b[0];
}

extern "C" void kernel_launch(void* const* d_in, const int* in_sizes, int n_in,
                              void* d_out, int out_size, void* d_ws, size_t ws_size,
                              hipStream_t stream) {
  const float* x     = (const float*)d_in[0];
  const int*   src   = (const int*)d_in[1];
  const int*   dst   = (const int*)d_in[2];
  const float* gcn_w = (const float*)d_in[3];
  const float* gcn_b = (const float*)d_in[4];
  const float* ff_w1 = (const float*)d_in[5];
  const float* ff_b1 = (const float*)d_in[6];
  const float* ln_g  = (const float*)d_in[7];
  const float* ln_b  = (const float*)d_in[8];
  const float* ff_w2 = (const float*)d_in[9];
  const float* ff_b2 = (const float*)d_in[10];
  const float* fc1_w = (const float*)d_in[11];
  const float* fc1_b = (const float*)d_in[12];
  const float* bn1_g = (const float*)d_in[13];
  const float* bn1_b = (const float*)d_in[14];
  const float* fc2_w = (const float*)d_in[15];
  const float* fc2_b = (const float*)d_in[16];
  const float* bn2_g = (const float*)d_in[17];
  const float* bn2_b = (const float*)d_in[18];
  const float* fc3_w = (const float*)d_in[19];
  const float* fc3_b = (const float*)d_in[20];
  const float* bn3_g = (const float*)d_in[21];
  const float* bn3_b = (const float*)d_in[22];
  const float* fc4_w = (const float*)d_in[23];
  const float* fc4_b = (const float*)d_in[24];
  float* out = (float*)d_out;

  const long N = in_sizes[0] / 128;
  const long E = in_sizes[1];

  char* wsp = (char*)d_ws;
  auto take = [&](size_t bytes) {
    char* p = wsp;
    wsp += (bytes + 255) & ~(size_t)255;
    return p;
  };
  float*    ro    = (float*)take(N * 4);
  float*    ri    = (float*)take(N * 4);
  float*    aggf  = (float*)take(N * 128 * 4);        // f32 scatter-add target
  _Float16* A1h   = (_Float16*)take(N * 128 * 2);
  _Float16* A2h   = (_Float16*)take(N * 128 * 2);
  _Float16* B1h   = (_Float16*)take(N * 512 * 2);
  _Float16* B2h   = (_Float16*)take(N * 512 * 2);
  float*    bns   = (float*)take(1024 * 4);
  _Float16* gcn_t = (_Float16*)take(128 * 128 * 2);   // [Kout][Kin] f16 weights
  _Float16* ff1_t = (_Float16*)take(512 * 128 * 2);
  _Float16* ff2_t = (_Float16*)take(128 * 512 * 2);
  _Float16* fc1_t = (_Float16*)take(512 * 128 * 2);
  _Float16* fc2_t = (_Float16*)take(512 * 512 * 2);
  _Float16* fc3_t = (_Float16*)take(128 * 512 * 2);

  const int TPB = 256;
  zero_f32<<<1024, TPB, 0, stream>>>(ro, N);
  zero_f32<<<1024, TPB, 0, stream>>>(ri, N);
  zero_f32<<<8192, TPB, 0, stream>>>(aggf, N * 128);

  // one-shot weight transpose/convert (L2-resident afterwards)
  wtrans_kernel<<<(128 * 128 + TPB - 1) / TPB, TPB, 0, stream>>>(gcn_w, gcn_t, 128, 128);
  wtrans_kernel<<<(512 * 128 + TPB - 1) / TPB, TPB, 0, stream>>>(ff_w1, ff1_t, 128, 512);
  wtrans_kernel<<<(128 * 512 + TPB - 1) / TPB, TPB, 0, stream>>>(ff_w2, ff2_t, 512, 128);
  wtrans_kernel<<<(512 * 128 + TPB - 1) / TPB, TPB, 0, stream>>>(fc1_w, fc1_t, 128, 512);
  wtrans_kernel<<<(512 * 512 + TPB - 1) / TPB, TPB, 0, stream>>>(fc2_w, fc2_t, 512, 512);
  wtrans_kernel<<<(128 * 512 + TPB - 1) / TPB, TPB, 0, stream>>>(fc3_w, fc3_t, 512, 128);

  degree_kernel<<<(int)((E + TPB - 1) / TPB), TPB, 0, stream>>>(src, dst, ro, ri, E);
  rsqrt_clip_kernel<<<(int)((N + TPB - 1) / TPB), TPB, 0, stream>>>(ro, N);
  rsqrt_clip_kernel<<<(int)((N + TPB - 1) / TPB), TPB, 0, stream>>>(ri, N);
  scale_rows_kernel<<<(int)((N * 128 + TPB - 1) / TPB), TPB, 0, stream>>>(x, ro, A1h, N * 128);
  scatter_kernel<<<(int)((E * 32 + TPB - 1) / TPB), TPB, 0, stream>>>(A1h, src, dst, aggf, E);
  cvt_f32_f16_kernel<<<(int)((N * 128 + TPB - 1) / TPB), TPB, 0, stream>>>(aggf, A2h, N * 128);

  dim3 blk(256);
  int mt = (int)((N + 127) / 128);
  // GraphConv: A1h = (A2h @ gcn_w) * ri[:,None] + gcn_b
  gemm_wmma_kernel<EPI_GCN><<<dim3(1, mt), blk, 0, stream>>>(A2h, gcn_t, gcn_b, ri, A1h, (int)N, 128, 128);
  // FF block
  gemm_wmma_kernel<EPI_GELU><<<dim3(4, mt), blk, 0, stream>>>(A1h, ff1_t, ff_b1, nullptr, B1h, (int)N, 128, 512);
  layernorm_kernel<<<(int)((N + 7) / 8), TPB, 0, stream>>>(B1h, ln_g, ln_b, (int)N);
  gemm_wmma_kernel<EPI_LIN><<<dim3(1, mt), blk, 0, stream>>>(B1h, ff2_t, ff_b2, nullptr, A2h, (int)N, 512, 128);
  // MLP head
  gemm_wmma_kernel<EPI_GELU><<<dim3(4, mt), blk, 0, stream>>>(A2h, fc1_t, fc1_b, nullptr, B1h, (int)N, 128, 512);
  zero_f32<<<4, TPB, 0, stream>>>(bns, 1024);
  bn_stats_kernel<<<(int)((N + 255) / 256), TPB, 0, stream>>>(B1h, bns, (int)N, 512);
  bn_apply_kernel<<<(int)((N * 512 + TPB - 1) / TPB), TPB, 0, stream>>>(B1h, bns, bn1_g, bn1_b, N, 512);

  gemm_wmma_kernel<EPI_GELU><<<dim3(4, mt), blk, 0, stream>>>(B1h, fc2_t, fc2_b, nullptr, B2h, (int)N, 512, 512);
  zero_f32<<<4, TPB, 0, stream>>>(bns, 1024);
  bn_stats_kernel<<<(int)((N + 255) / 256), TPB, 0, stream>>>(B2h, bns, (int)N, 512);
  bn_apply_kernel<<<(int)((N * 512 + TPB - 1) / TPB), TPB, 0, stream>>>(B2h, bns, bn2_g, bn2_b, N, 512);

  gemm_wmma_kernel<EPI_GELU><<<dim3(1, mt), blk, 0, stream>>>(B2h, fc3_t, fc3_b, nullptr, A1h, (int)N, 512, 128);
  zero_f32<<<4, TPB, 0, stream>>>(bns, 1024);
  bn_stats_kernel<<<(int)((N + 255) / 256), TPB, 0, stream>>>(A1h, bns, (int)N, 128);
  bn_apply_kernel<<<(int)((N * 128 + TPB - 1) / TPB), TPB, 0, stream>>>(A1h, bns, bn3_g, bn3_b, N, 128);

  fc4_kernel<<<(int)((N + 7) / 8), TPB, 0, stream>>>(A1h, fc4_w, fc4_b, out, (int)N);
}
